// Attention_47802986004693
// MI455X (gfx1250) — compile-verified, matrix operand order checked by reading, not run on hardware
//
#include <hip/hip_runtime.h>
#include <hip/hip_bf16.h>

typedef __attribute__((ext_vector_type(16))) __bf16 v16bf;
typedef __attribute__((ext_vector_type(8)))  __bf16 v8bf;
typedef __attribute__((ext_vector_type(2)))  __bf16 v2bf;
typedef __attribute__((ext_vector_type(8)))  float  v8f;
typedef int v4i_vs __attribute__((vector_size(16)));   // matches builtin prototype

#define BB   2
#define SS   2048
#define DD   1024
#define HH   16
#define HDD  64
#define MM   (BB * SS)      // 4096 rows for the projection GEMMs
#define LDP  40             // padded LDS row length (bf16 elems)
#define MEG  (1u << 20)     // one weight matrix = 1M elements

// ---- feature gates (compile-safe probes) ---------------------------------
#if defined(__gfx1250__) && __has_builtin(__builtin_amdgcn_global_load_async_to_lds_b128) && \
    __has_builtin(__builtin_amdgcn_s_wait_asynccnt)
#define USE_ASYNC_LDS 1
#endif

#if defined(__gfx1250__) && __has_builtin(__builtin_amdgcn_cvt_pk_bf16_f32)
#define USE_PK_BF16 1
#endif

// ---- helpers -------------------------------------------------------------

__device__ __forceinline__ __bf16 f2bf(float f) {
  unsigned u = __builtin_bit_cast(unsigned, f);
  u += 0x7fffu + ((u >> 16) & 1u);
  unsigned short h = (unsigned short)(u >> 16);
  return __builtin_bit_cast(__bf16, h);
}

__device__ __forceinline__ v8bf cvt8(float4 a, float4 b) {
  v8bf r;
#ifdef USE_PK_BF16
  v2bf p0 = __builtin_amdgcn_cvt_pk_bf16_f32(a.x, a.y);
  v2bf p1 = __builtin_amdgcn_cvt_pk_bf16_f32(a.z, a.w);
  v2bf p2 = __builtin_amdgcn_cvt_pk_bf16_f32(b.x, b.y);
  v2bf p3 = __builtin_amdgcn_cvt_pk_bf16_f32(b.z, b.w);
  r[0] = p0[0]; r[1] = p0[1]; r[2] = p1[0]; r[3] = p1[1];
  r[4] = p2[0]; r[5] = p2[1]; r[6] = p3[0]; r[7] = p3[1];
#else
  r[0] = f2bf(a.x); r[1] = f2bf(a.y); r[2] = f2bf(a.z); r[3] = f2bf(a.w);
  r[4] = f2bf(b.x); r[5] = f2bf(b.y); r[6] = f2bf(b.z); r[7] = f2bf(b.w);
#endif
  return r;
}

__device__ __forceinline__ v16bf cat16(v8bf lo, v8bf hi) {
  return __builtin_shufflevector(lo, hi, 0, 1, 2, 3, 4, 5, 6, 7,
                                 8, 9, 10, 11, 12, 13, 14, 15);
}

#ifdef USE_ASYNC_LDS
// 16-byte global -> LDS async copy (ASYNCcnt-tracked, bypasses VGPRs).
__device__ __forceinline__ void async_cp16(const __bf16* g, __bf16* l) {
  __builtin_amdgcn_global_load_async_to_lds_b128(
      (__attribute__((address_space(1))) v4i_vs*)(unsigned long long)(uintptr_t)g,
      (__attribute__((address_space(3))) v4i_vs*)(unsigned)(uintptr_t)l,
      0, 0);
}
#endif

// ==========================================================================
// Kernel 0: one-shot fp32 -> bf16 pre-conversion of x and the 4 weights.
// Layout in dst: [ xb: 4M | qw: 1M | kw: 1M | vw: 1M | ow: 1M ]
// ==========================================================================
__global__ __launch_bounds__(256)
void cvt_bf16_kernel(const float* __restrict__ x,
                     const float* __restrict__ qw, const float* __restrict__ kw,
                     const float* __restrict__ vw, const float* __restrict__ ow,
                     __bf16* __restrict__ dst) {
  const size_t p = ((size_t)blockIdx.x * 256 + threadIdx.x) * 8;
  const float* src;
  if (p < (size_t)MM * DD) {
    src = x + p;
  } else {
    const size_t q = p - (size_t)MM * DD;
    const int which = (int)(q >> 20);
    const float* w = (which == 0) ? qw : (which == 1) ? kw : (which == 2) ? vw : ow;
    src = w + (q & (MEG - 1));
  }
  float4 a = ((const float4*)src)[0];
  float4 b = ((const float4*)src)[1];
  *(v8bf*)(dst + p) = cvt8(a, b);
}

// ==========================================================================
// Kernel 1: fused Q/K/V projection, 128x64 tile per 128-thr block, K-step 32,
// LDS ping-pong double buffering with async loads. Each wave: 32x64 strip,
// 2 A-fragments x 4 B-fragments = 8 WMMAs per K-step (B reused 2x).
// grid = (M/128, D/64, 3)
// ==========================================================================
__global__ __launch_bounds__(128)
void qkv_proj_kernel(const __bf16* __restrict__ xb,
                     const __bf16* __restrict__ wq, const __bf16* __restrict__ wk,
                     const __bf16* __restrict__ wv,
                     const float* __restrict__ qb, const float* __restrict__ kb,
                     const float* __restrict__ vb,
                     __bf16* __restrict__ qkv_ws) {
  __shared__ __attribute__((aligned(16))) __bf16 As[2][128][LDP];
  __shared__ __attribute__((aligned(16))) __bf16 Bs[2][64][LDP];

  const int which = blockIdx.z;
  const __bf16* W   = (which == 0) ? wq : (which == 1) ? wk : wv;
  const float* bias = (which == 0) ? qb : (which == 1) ? kb : vb;
  __bf16* dst = qkv_ws + (size_t)which * (size_t)MM * DD;

  const int m0   = blockIdx.x * 128;
  const int n0   = blockIdx.y * 64;
  const int tid  = threadIdx.x;
  const int lane = tid & 31;
  const int w    = tid >> 5;
  const int l16  = lane & 15;
  const int lhi  = lane >> 4;
  const int kc0  = lhi * 8;            // fragment K-chunk base (ISA 16-bit layout)

  const int brow_s = tid >> 1;         // B staging row 0..63
  const int bcol_s = (tid & 1) * 16;   // B staging col base 0|16

  // stage one 128x32 A tile (row per thread) + 64x32 B tile (half row/thread)
  auto stage = [&](int buf, int k0) {
    const __bf16* ap = xb + (size_t)(m0 + tid) * DD + k0;
    const __bf16* bp = W  + (size_t)(n0 + brow_s) * DD + k0 + bcol_s;
#ifdef USE_ASYNC_LDS
#pragma unroll
    for (int j = 0; j < 4; ++j) async_cp16(ap + j * 8, &As[buf][tid][j * 8]);
    async_cp16(bp,     &Bs[buf][brow_s][bcol_s]);
    async_cp16(bp + 8, &Bs[buf][brow_s][bcol_s + 8]);
#else
#pragma unroll
    for (int j = 0; j < 4; ++j)
      *(v8bf*)&As[buf][tid][j * 8] = *(const v8bf*)(ap + j * 8);
    *(v8bf*)&Bs[buf][brow_s][bcol_s]     = *(const v8bf*)bp;
    *(v8bf*)&Bs[buf][brow_s][bcol_s + 8] = *(const v8bf*)(bp + 8);
#endif
  };

  v8f acc[2][4] = {};

  stage(0, 0);
#ifdef USE_ASYNC_LDS
  __builtin_amdgcn_s_wait_asynccnt(0);
#endif
  __syncthreads();

  for (int k0 = 0; k0 < DD; k0 += 32) {
    const int cur = (k0 >> 5) & 1;
    if (k0 + 32 < DD) stage(cur ^ 1, k0 + 32);   // prefetch overlaps compute

    v16bf af[2];
#pragma unroll
    for (int i = 0; i < 2; ++i) {
      const int arow = w * 32 + i * 16 + l16;
      af[i] = cat16(*(const v8bf*)&As[cur][arow][kc0],
                    *(const v8bf*)&As[cur][arow][kc0 + 16]);
    }
#pragma unroll
    for (int t = 0; t < 4; ++t) {
      const int brow = t * 16 + l16;
      v16bf bf = cat16(*(const v8bf*)&Bs[cur][brow][kc0],
                       *(const v8bf*)&Bs[cur][brow][kc0 + 16]);
#pragma unroll
      for (int i = 0; i < 2; ++i)
        acc[i][t] = __builtin_amdgcn_wmma_f32_16x16x32_bf16(
            false, af[i], false, bf, (short)0, acc[i][t], false, false);
    }

#ifdef USE_ASYNC_LDS
    __builtin_amdgcn_s_wait_asynccnt(0);         // next tile landed
#endif
    __syncthreads();                             // includes s_wait_dscnt 0
  }

#pragma unroll
  for (int t = 0; t < 4; ++t) {
    const int ncol = n0 + t * 16 + l16;
    const float bv = bias[ncol];
    const int h = ncol >> 6, hd = ncol & 63;
#pragma unroll
    for (int i = 0; i < 2; ++i) {
#pragma unroll
      for (int r = 0; r < 8; ++r) {
        const int m = m0 + w * 32 + i * 16 + r + (lhi << 3);
        const int bb = m >> 11, s = m & (SS - 1);
        dst[((size_t)(bb * HH + h) * SS + s) * HDD + hd] = f2bf(acc[i][t][r] + bv);
      }
    }
  }
}

// ==========================================================================
// Kernel 2: causal flash attention — WMMA QK^T / PV, online softmax in log2
// domain, V transposed via LDS, P (C-layout) -> A-fragment via LDS bounce.
// grid = (S/64, B*H)
// ==========================================================================
__global__ __launch_bounds__(128)
void attn_kernel(const __bf16* __restrict__ qkv_ws, __bf16* __restrict__ o_ws) {
  __shared__ __attribute__((aligned(16))) __bf16 Vt[64][LDP];       // [hd][key]
  __shared__ __attribute__((aligned(16))) __bf16 Pst[4][16][LDP];   // per-wave P bounce

  const __bf16* q_ws = qkv_ws;
  const __bf16* k_ws = qkv_ws + (size_t)MM * DD;
  const __bf16* v_ws = qkv_ws + 2ull * MM * DD;

  const int bh = blockIdx.y;
  const size_t base = (size_t)bh * SS * HDD;
  const int qb0  = blockIdx.x * 64;
  const int tid  = threadIdx.x;
  const int lane = tid & 31;
  const int w    = tid >> 5;
  const int wqb  = qb0 + w * 16;
  const int l16  = lane & 15;
  const int lhi  = lane >> 4;
  const int kc0  = lhi * 8;

  const __bf16* qp = q_ws + base + (size_t)(wqb + l16) * HDD;
  v16bf qf0 = cat16(*(const v8bf*)(qp + kc0),      *(const v8bf*)(qp + kc0 + 16));
  v16bf qf1 = cat16(*(const v8bf*)(qp + 32 + kc0), *(const v8bf*)(qp + 32 + kc0 + 16));

  v8f oacc[4] = {};
  float mrow[8], lrow[8];
#pragma unroll
  for (int r = 0; r < 8; ++r) { mrow[r] = -1e30f; lrow[r] = 0.f; }

  const float ksc = 0.125f * 1.44269504088896f;      // 1/sqrt(64) * log2(e)
  const int nkb = 2 * blockIdx.x + 2;

  const int skey = tid >> 2;
  const int shd  = (tid & 3) * 16;

  for (int kb = 0; kb < nkb; ++kb) {
    __syncthreads();
    {
      const __bf16* vp = v_ws + base + (size_t)(kb * 32 + skey) * HDD + shd;
      v8bf a = *(const v8bf*)vp;
      v8bf b = *(const v8bf*)(vp + 8);
#pragma unroll
      for (int i = 0; i < 8; ++i) {
        Vt[shd + i][skey]     = a[i];
        Vt[shd + 8 + i][skey] = b[i];
      }
    }
    __syncthreads();

    if (kb * 32 <= wqb + 15) {                       // wave-uniform causal skip
      float ps[2][8];
#pragma unroll
      for (int t = 0; t < 2; ++t) {
        const int key = kb * 32 + t * 16 + l16;
        const __bf16* kp = k_ws + base + (size_t)key * HDD;
        v16bf kf0 = cat16(*(const v8bf*)(kp + kc0),      *(const v8bf*)(kp + kc0 + 16));
        v16bf kf1 = cat16(*(const v8bf*)(kp + 32 + kc0), *(const v8bf*)(kp + 32 + kc0 + 16));
        v8f sa = {};
        sa = __builtin_amdgcn_wmma_f32_16x16x32_bf16(false, qf0, false, kf0, (short)0, sa, false, false);
        sa = __builtin_amdgcn_wmma_f32_16x16x32_bf16(false, qf1, false, kf1, (short)0, sa, false, false);
#pragma unroll
        for (int r = 0; r < 8; ++r) {
          const int qrow = wqb + r + (lhi << 3);
          float s = sa[r] * ksc;
          if (key > qrow) s = -1e30f;
          ps[t][r] = s;
        }
      }
#pragma unroll
      for (int r = 0; r < 8; ++r) {
        float mx = fmaxf(ps[0][r], ps[1][r]);
#pragma unroll
        for (int msk = 1; msk < 16; msk <<= 1)
          mx = fmaxf(mx, __shfl_xor(mx, msk, 32));
        const float mnew = fmaxf(mrow[r], mx);
        const float corr = __builtin_exp2f(mrow[r] - mnew);
        const float p0 = __builtin_exp2f(ps[0][r] - mnew);
        const float p1 = __builtin_exp2f(ps[1][r] - mnew);
        float rs = p0 + p1;
#pragma unroll
        for (int msk = 1; msk < 16; msk <<= 1)
          rs += __shfl_xor(rs, msk, 32);
        lrow[r] = lrow[r] * corr + rs;
        mrow[r] = mnew;
#pragma unroll
        for (int t = 0; t < 4; ++t) oacc[t][r] = oacc[t][r] * corr;
        Pst[w][r + (lhi << 3)][l16]      = f2bf(p0);
        Pst[w][r + (lhi << 3)][16 + l16] = f2bf(p1);
      }
      v16bf pf = cat16(*(const v8bf*)&Pst[w][l16][kc0],
                       *(const v8bf*)&Pst[w][l16][kc0 + 16]);
#pragma unroll
      for (int t = 0; t < 4; ++t) {
        const int hd = t * 16 + l16;
        v16bf vf = cat16(*(const v8bf*)&Vt[hd][kc0],
                         *(const v8bf*)&Vt[hd][kc0 + 16]);
        oacc[t] = __builtin_amdgcn_wmma_f32_16x16x32_bf16(
            false, pf, false, vf, (short)0, oacc[t], false, false);
      }
    }
  }

#pragma unroll
  for (int r = 0; r < 8; ++r) {
    const float inv = 1.0f / lrow[r];
    const int s = wqb + r + (lhi << 3);
#pragma unroll
    for (int t = 0; t < 4; ++t)
      o_ws[base + (size_t)s * HDD + t * 16 + l16] = f2bf(oacc[t][r] * inv);
  }
}

// ==========================================================================
// Kernel 3: output projection, same 128x64 double-buffered tiling; A read
// bf16 from [B,H,S,HD] workspace; fp32 result + bias.
// grid = (M/128, D/64)
// ==========================================================================
__global__ __launch_bounds__(128)
void out_proj_kernel(const __bf16* __restrict__ o_ws,
                     const __bf16* __restrict__ wo, const float* __restrict__ ob,
                     float* __restrict__ out) {
  __shared__ __attribute__((aligned(16))) __bf16 As[2][128][LDP];
  __shared__ __attribute__((aligned(16))) __bf16 Bs[2][64][LDP];

  const int m0   = blockIdx.x * 128;
  const int n0   = blockIdx.y * 64;
  const int tid  = threadIdx.x;
  const int lane = tid & 31;
  const int w    = tid >> 5;
  const int l16  = lane & 15;
  const int lhi  = lane >> 4;
  const int kc0  = lhi * 8;

  const int brow_s = tid >> 1;
  const int bcol_s = (tid & 1) * 16;

  const int ma = m0 + tid;                    // this thread's A staging row
  const int bba = ma >> 11, sa = ma & (SS - 1);

  auto stage = [&](int buf, int k0) {
    // 32-col chunk stays inside one head (HD=64, k0 multiple of 32)
    const int h = k0 >> 6, hd = k0 & 63;
    const __bf16* ap = o_ws + ((size_t)(bba * HH + h) * SS + sa) * HDD + hd;
    const __bf16* bp = wo + (size_t)(n0 + brow_s) * DD + k0 + bcol_s;
#ifdef USE_ASYNC_LDS
#pragma unroll
    for (int j = 0; j < 4; ++j) async_cp16(ap + j * 8, &As[buf][tid][j * 8]);
    async_cp16(bp,     &Bs[buf][brow_s][bcol_s]);
    async_cp16(bp + 8, &Bs[buf][brow_s][bcol_s + 8]);
#else
#pragma unroll
    for (int j = 0; j < 4; ++j)
      *(v8bf*)&As[buf][tid][j * 8] = *(const v8bf*)(ap + j * 8);
    *(v8bf*)&Bs[buf][brow_s][bcol_s]     = *(const v8bf*)bp;
    *(v8bf*)&Bs[buf][brow_s][bcol_s + 8] = *(const v8bf*)(bp + 8);
#endif
  };

  v8f acc[2][4] = {};

  stage(0, 0);
#ifdef USE_ASYNC_LDS
  __builtin_amdgcn_s_wait_asynccnt(0);
#endif
  __syncthreads();

  for (int k0 = 0; k0 < DD; k0 += 32) {
    const int cur = (k0 >> 5) & 1;
    if (k0 + 32 < DD) stage(cur ^ 1, k0 + 32);

    v16bf af[2];
#pragma unroll
    for (int i = 0; i < 2; ++i) {
      const int arow = w * 32 + i * 16 + l16;
      af[i] = cat16(*(const v8bf*)&As[cur][arow][kc0],
                    *(const v8bf*)&As[cur][arow][kc0 + 16]);
    }
#pragma unroll
    for (int t = 0; t < 4; ++t) {
      const int brow = t * 16 + l16;
      v16bf bf = cat16(*(const v8bf*)&Bs[cur][brow][kc0],
                       *(const v8bf*)&Bs[cur][brow][kc0 + 16]);
#pragma unroll
      for (int i = 0; i < 2; ++i)
        acc[i][t] = __builtin_amdgcn_wmma_f32_16x16x32_bf16(
            false, af[i], false, bf, (short)0, acc[i][t], false, false);
    }

#ifdef USE_ASYNC_LDS
    __builtin_amdgcn_s_wait_asynccnt(0);
#endif
    __syncthreads();
  }

#pragma unroll
  for (int t = 0; t < 4; ++t) {
    const int ncol = n0 + t * 16 + l16;
    const float bv = ob[ncol];
#pragma unroll
    for (int i = 0; i < 2; ++i) {
#pragma unroll
      for (int r = 0; r < 8; ++r) {
        const int m = m0 + w * 32 + i * 16 + r + (lhi << 3);
        out[(size_t)m * DD + ncol] = acc[i][t][r] + bv;
      }
    }
  }
}

// ==========================================================================
extern "C" void kernel_launch(void* const* d_in, const int* in_sizes, int n_in,
                              void* d_out, int out_size, void* d_ws, size_t ws_size,
                              hipStream_t stream) {
  const float* x  = (const float*)d_in[0];
  const float* qw = (const float*)d_in[1];
  const float* qb = (const float*)d_in[2];
  const float* kw = (const float*)d_in[3];
  const float* kb = (const float*)d_in[4];
  const float* vw = (const float*)d_in[5];
  const float* vb = (const float*)d_in[6];
  const float* ow = (const float*)d_in[7];
  const float* ob = (const float*)d_in[8];

  // bf16 workspace layout (elements):
  // [ xb 4M | wq 1M | wk 1M | wv 1M | wo 1M | qkv 12M | attn-out 4M ] = 48 MB
  __bf16* ws  = (__bf16*)d_ws;
  __bf16* xb  = ws;
  __bf16* wqb = ws + 4ull * MEG;
  __bf16* wkb = ws + 5ull * MEG;
  __bf16* wvb = ws + 6ull * MEG;
  __bf16* wob = ws + 7ull * MEG;
  __bf16* qkv = ws + 8ull * MEG;
  __bf16* oat = ws + 20ull * MEG;

  // 0) one-shot bf16 conversion of x + all weights (8M elems, 8/thread)
  cvt_bf16_kernel<<<dim3(4096), 256, 0, stream>>>(x, qw, kw, vw, ow, xb);

  // 1) fused QKV projection
  dim3 g1(MM / 128, DD / 64, 3);
  qkv_proj_kernel<<<g1, 128, 0, stream>>>(xb, wqb, wkb, wvb, qb, kb, vb, qkv);

  // 2) causal flash attention
  dim3 g2(SS / 64, BB * HH);
  attn_kernel<<<g2, 128, 0, stream>>>(qkv, oat);

  // 3) output projection (fp32 result)
  dim3 g3(MM / 128, DD / 64);
  out_proj_kernel<<<g3, 128, 0, stream>>>(oat, wob, ob, (float*)d_out);
}
